// EncoderDecoder_16045997818341
// MI455X (gfx1250) — matrix operand order repaired and checked by reading, not compiled, not run on hardware
//
#include <hip/hip_runtime.h>

// ---------------------------------------------------------------------------
// LSTM encoder-decoder for MI455X (gfx1250), bf16 WMMA with fp32 accumulate.
// B=512, T=512, I=128, H=512. One kernel launch per timestep (1024 steps),
// fused input+recurrent GEMM (K = I + H = 640), in-register gate activation
// and (h, c) update. hbf (bf16 h for next step's GEMM) is double-buffered.
// If workspace allows, x is pre-converted to bf16 once per call (removes the
// v_cvt_pk_bf16_f32 chain from the per-step critical path and halves x BW).
// ---------------------------------------------------------------------------

typedef __bf16 bhalf;
typedef __attribute__((ext_vector_type(16))) __bf16 v16bf;
typedef __attribute__((ext_vector_type(8)))  __bf16 v8bf;
typedef __attribute__((ext_vector_type(8)))  float  v8f;

#define BB   512   // batch
#define TT   512   // timesteps
#define II   128   // input dim
#define HH   512   // hidden dim
#define KK   (II + HH)     // 640, fused GEMM K
#define G4H  (4 * HH)      // 2048 gate columns

__device__ __forceinline__ float sigm(float x) {
    return 1.0f / (1.0f + __expf(-x));
}
__device__ __forceinline__ float tanh_fast(float x) {
    float e = __expf(-2.0f * fabsf(x));
    float t = (1.0f - e) / (1.0f + e);
    return x >= 0.0f ? t : -t;
}

// ---- Pack [Wih | Whh] (fp32) -> bf16, row-major [4H][KK] ------------------
__global__ void pack_weights_kernel(const float* __restrict__ Wih,
                                    const float* __restrict__ Whh,
                                    bhalf* __restrict__ Wc) {
    int idx = blockIdx.x * blockDim.x + threadIdx.x;     // over 4H*KK
    if (idx >= G4H * KK) return;
    int g = idx / KK;
    int k = idx - g * KK;
    float v = (k < II) ? Wih[g * II + k] : Whh[g * HH + (k - II)];
    Wc[idx] = (bhalf)v;
}

// ---- bias = bih + bhh (fp32, 4H) ------------------------------------------
__global__ void pack_bias_kernel(const float* __restrict__ bih,
                                 const float* __restrict__ bhh,
                                 float* __restrict__ bias) {
    int idx = blockIdx.x * blockDim.x + threadIdx.x;
    if (idx < G4H) bias[idx] = bih[idx] + bhh[idx];
}

// ---- zero c and both bf16 h buffers ---------------------------------------
__global__ void zero_state_kernel(float* __restrict__ c,
                                  bhalf* __restrict__ hbf0,
                                  bhalf* __restrict__ hbf1) {
    int idx = blockIdx.x * blockDim.x + threadIdx.x;     // over B*H
    if (idx >= BB * HH) return;
    c[idx] = 0.0f;
    hbf0[idx] = (bhalf)0.0f;
    hbf1[idx] = (bhalf)0.0f;
}

// ---- one-shot x fp32 -> bf16 (4 elems/thread, vectorized) -----------------
__global__ void convert_x_kernel(const float* __restrict__ x,
                                 bhalf* __restrict__ xbf) {
    const int n4 = (BB * TT * II) / 4;                   // 8,388,608
    int i = blockIdx.x * blockDim.x + threadIdx.x;
    if (i >= n4) return;
    const float4 f = *(const float4*)(x + (size_t)i * 4);
    v8bf* dst = (v8bf*)xbf;                              // write 8B chunks
    struct B4 { bhalf a, b, c, d; };
    B4 o = { (bhalf)f.x, (bhalf)f.y, (bhalf)f.z, (bhalf)f.w };
    *(B4*)((bhalf*)dst + (size_t)i * 4) = o;
}

// ---------------------------------------------------------------------------
// One LSTM timestep:
//   gates[b, g] = sum_k [x_t | h][b, k] * Wc[g, k]       (bf16 WMMA, f32 acc)
//   i,f,g,o -> c_new = f*c + i*g ; h_new = o*tanh(c_new)
// Grid: (B/64, H/32) blocks of 256 threads (8 wave32).
// Wave (rt, nt) owns a 16x16 (batch, h) tile and all four gates for it.
// XBF=true: x already bf16 (xbf); XBF=false: convert x fp32 on the fly.
// ---------------------------------------------------------------------------
template <bool XBF>
__global__ __launch_bounds__(256) void lstm_step_kernel(
        const float* __restrict__ x,       // [B][T][I] fp32 (XBF=false)
        const bhalf* __restrict__ xbf,     // [B][T][I] bf16 (XBF=true)
        int t,
        const bhalf* __restrict__ Wc,      // [4H][KK] bf16
        const float* __restrict__ bias,    // [4H]
        float* __restrict__ h,             // [B][H] fp32 (write-only here)
        float* __restrict__ c,             // [B][H] fp32 (read+write, owned)
        const bhalf* __restrict__ hread,   // [B][H] bf16 (prev step's h)
        bhalf* __restrict__ hwrite)        // [B][H] bf16 (this step's h)
{
    const int lane = threadIdx.x & 31;
    const int wv   = threadIdx.x >> 5;   // 0..7
    const int rt   = wv & 3;             // row tile   (4 x 16 batch rows)
    const int nt   = wv >> 2;            // col tile   (2 x 16 h cols)
    const int half = lane >> 4;          // 0/1: lane half
    const int m    = lane & 15;

    const int b_base = blockIdx.x * 64 + rt * 16;   // batch rows of this wave
    const int h_base = blockIdx.y * 32 + nt * 16;   // h cols of this wave

    // Accumulators: one per gate (i, f, g, o), fp32 16x16.
    v8f acc[4];
    const v8f z8 = {0.f, 0.f, 0.f, 0.f, 0.f, 0.f, 0.f, 0.f};
#pragma unroll
    for (int q = 0; q < 4; ++q) acc[q] = z8;

    // B-matrix row pointers: column n of B == row (q*H + h_base + m) of Wc.
    const bhalf* wrow[4];
#pragma unroll
    for (int q = 0; q < 4; ++q)
        wrow[q] = Wc + (size_t)(q * HH + h_base + m) * KK;

    // A-matrix row for this lane (both lane halves hold the same row).
    const int arow = b_base + m;
    const bhalf* hrow = hread + (size_t)arow * HH;

    union AFrag { v16bf v; v8bf half8[2]; };

    // ---- Phase 1: K over x_t, k = 0..127 -----------------------------------
    if constexpr (XBF) {
        const bhalf* xrow = xbf + ((size_t)arow * TT + t) * II;
#pragma unroll
        for (int k0 = 0; k0 < II; k0 += 32) {
            AFrag a;
            a.half8[0] = *(const v8bf*)(xrow + k0 + half * 8);
            a.half8[1] = *(const v8bf*)(xrow + k0 + 16 + half * 8);
#pragma unroll
            for (int q = 0; q < 4; ++q) {
                AFrag bf_; bf_.half8[0] = *(const v8bf*)(wrow[q] + k0 + half * 8); bf_.half8[1] = *(const v8bf*)(wrow[q] + k0 + 16 + half * 8); v16bf bb = bf_.v;
                acc[q] = __builtin_amdgcn_wmma_f32_16x16x32_bf16(
                             false, a.v, false, bb, (short)0, acc[q], false, false);
            }
        }
    } else {
        const float* xrow = x + ((size_t)arow * TT + t) * II;
        for (int k0 = 0; k0 < II; k0 += 32) {
            const float* p1 = xrow + k0 + half * 8;    // K = k0 + half*8 ..+7
            const float* p2 = p1 + 16;                 // K = k0+16+half*8..+7
            AFrag a;
#pragma unroll
            for (int i = 0; i < 8; ++i) {
                a.v[i]     = (bhalf)p1[i];
                a.v[8 + i] = (bhalf)p2[i];
            }
#pragma unroll
            for (int q = 0; q < 4; ++q) {
                AFrag bf_; bf_.half8[0] = *(const v8bf*)(wrow[q] + k0 + half * 8); bf_.half8[1] = *(const v8bf*)(wrow[q] + k0 + 16 + half * 8); v16bf bb = bf_.v;
                acc[q] = __builtin_amdgcn_wmma_f32_16x16x32_bf16(
                             false, a.v, false, bb, (short)0, acc[q], false, false);
            }
        }
    }

    // ---- Phase 2: K over h_{t-1} (bf16), k = 128..639 ----------------------
#pragma unroll 2
    for (int k0 = 0; k0 < HH; k0 += 32) {
        AFrag a;
        a.half8[0] = *(const v8bf*)(hrow + k0 + half * 8);
        a.half8[1] = *(const v8bf*)(hrow + k0 + 16 + half * 8);
#pragma unroll
        for (int q = 0; q < 4; ++q) {
            AFrag bf_; bf_.half8[0] = *(const v8bf*)(wrow[q] + II + k0 + half * 8); bf_.half8[1] = *(const v8bf*)(wrow[q] + II + k0 + 16 + half * 8); v16bf bb = bf_.v;
            acc[q] = __builtin_amdgcn_wmma_f32_16x16x32_bf16(
                         false, a.v, false, bb, (short)0, acc[q], false, false);
        }
    }

    // ---- Gate activation + state update (all in-register) -----------------
    // C/D layout: VGPR v, lanes 0-15 -> (M=v, N=lane); lanes 16-31 -> M=v+8.
    const int hcol = h_base + m;
    const float bi = bias[0 * HH + hcol];
    const float bf = bias[1 * HH + hcol];
    const float bg = bias[2 * HH + hcol];
    const float bo = bias[3 * HH + hcol];

#pragma unroll
    for (int v = 0; v < 8; ++v) {
        const int brow = b_base + half * 8 + v;
        const size_t idx = (size_t)brow * HH + hcol;
        float iv = sigm(acc[0][v] + bi);
        float fv = sigm(acc[1][v] + bf);
        float gv = tanh_fast(acc[2][v] + bg);
        float ov = sigm(acc[3][v] + bo);
        float cn = fv * c[idx] + iv * gv;
        float hn = ov * tanh_fast(cn);
        c[idx] = cn;
        h[idx] = hn;
        hwrite[idx] = (bhalf)hn;
    }
}

// ---- Final FC: out[b] = h[b,:] . fc_W + fc_b ------------------------------
__global__ __launch_bounds__(256) void fc_kernel(const float* __restrict__ h,
                                                 const float* __restrict__ fcW,
                                                 const float* __restrict__ fcb,
                                                 float* __restrict__ out) {
    const int lane = threadIdx.x & 31;
    const int wv   = threadIdx.x >> 5;
    const int b    = blockIdx.x * 8 + wv;      // 64 blocks * 8 waves = 512
    float s = 0.0f;
    for (int k = lane; k < HH; k += 32) s += h[(size_t)b * HH + k] * fcW[k];
#pragma unroll
    for (int off = 16; off > 0; off >>= 1) s += __shfl_down(s, off, 32);
    if (lane == 0) out[b] = s + fcb[0];
}

// ---------------------------------------------------------------------------
extern "C" void kernel_launch(void* const* d_in, const int* in_sizes, int n_in,
                              void* d_out, int out_size, void* d_ws, size_t ws_size,
                              hipStream_t stream) {
    const float* x       = (const float*)d_in[0];
    const float* enc_Wih = (const float*)d_in[1];
    const float* enc_Whh = (const float*)d_in[2];
    const float* enc_bih = (const float*)d_in[3];
    const float* enc_bhh = (const float*)d_in[4];
    const float* dec_Wih = (const float*)d_in[5];
    const float* dec_Whh = (const float*)d_in[6];
    const float* dec_bih = (const float*)d_in[7];
    const float* dec_bhh = (const float*)d_in[8];
    const float* fc_W    = (const float*)d_in[9];
    const float* fc_b    = (const float*)d_in[10];
    float* out = (float*)d_out;

    // Workspace carve-out (256B aligned).
    char* ws = (char*)d_ws;
    size_t off = 0;
    auto carve = [&](size_t bytes) -> void* {
        off = (off + 255) & ~(size_t)255;
        void* p = ws + off;
        off += bytes;
        return p;
    };
    bhalf* encW  = (bhalf*)carve((size_t)G4H * KK * sizeof(bhalf));
    bhalf* decW  = (bhalf*)carve((size_t)G4H * KK * sizeof(bhalf));
    float* encB  = (float*)carve((size_t)G4H * sizeof(float));
    float* decB  = (float*)carve((size_t)G4H * sizeof(float));
    float* hbuf  = (float*)carve((size_t)BB * HH * sizeof(float));
    float* cbuf  = (float*)carve((size_t)BB * HH * sizeof(float));
    bhalf* hbf0  = (bhalf*)carve((size_t)BB * HH * sizeof(bhalf));
    bhalf* hbf1  = (bhalf*)carve((size_t)BB * HH * sizeof(bhalf));
    // Optional bf16 copy of x (67 MB) — only if scratch is big enough.
    size_t xbf_off = (off + 255) & ~(size_t)255;
    const size_t xbf_bytes = (size_t)BB * TT * II * sizeof(bhalf);
    const bool use_xbf = (ws_size >= xbf_off + xbf_bytes);
    bhalf* xbf = use_xbf ? (bhalf*)(ws + xbf_off) : nullptr;
    (void)in_sizes; (void)n_in; (void)out_size;

    // --- Prep (idempotent, every call) ---
    {
        const int nW = G4H * KK;
        pack_weights_kernel<<<(nW + 255) / 256, 256, 0, stream>>>(enc_Wih, enc_Whh, encW);
        pack_weights_kernel<<<(nW + 255) / 256, 256, 0, stream>>>(dec_Wih, dec_Whh, decW);
        pack_bias_kernel<<<(G4H + 255) / 256, 256, 0, stream>>>(enc_bih, enc_bhh, encB);
        pack_bias_kernel<<<(G4H + 255) / 256, 256, 0, stream>>>(dec_bih, dec_bhh, decB);
        zero_state_kernel<<<(BB * HH + 255) / 256, 256, 0, stream>>>(cbuf, hbf0, hbf1);
        if (use_xbf) {
            const int n4 = (BB * TT * II) / 4;
            convert_x_kernel<<<(n4 + 255) / 256, 256, 0, stream>>>(x, xbf);
        }
    }

    // --- Encoder scan (global steps 0..511), then decoder (512..1023).
    // Step s reads hbf[s&1], writes hbf[(s+1)&1]. Decoder continues from
    // encoder's final (h, c) — exactly the reference semantics.
    const dim3 grid(BB / 64, HH / 32);   // 8 x 16 blocks
    for (int s = 0; s < 2 * TT; ++s) {
        const int t = (s < TT) ? s : (s - TT);
        const bhalf* Wc  = (s < TT) ? encW : decW;
        const float* bia = (s < TT) ? encB : decB;
        bhalf* hr = (s & 1) ? hbf1 : hbf0;
        bhalf* hw = (s & 1) ? hbf0 : hbf1;
        if (use_xbf) {
            lstm_step_kernel<true><<<grid, 256, 0, stream>>>(
                x, xbf, t, Wc, bia, hbuf, cbuf, hr, hw);
        } else {
            lstm_step_kernel<false><<<grid, 256, 0, stream>>>(
                x, xbf, t, Wc, bia, hbuf, cbuf, hr, hw);
        }
    }

    // --- Final FC on the last decoder hidden state ---
    fc_kernel<<<BB / 8, 256, 0, stream>>>(hbuf, fc_W, fc_b, out);
}
